// graphNetwork_seq_57389353009179
// MI455X (gfx1250) — matrix operand order repaired and checked by reading, not compiled, not run on hardware
//
#include <hip/hip_runtime.h>
#include <hip/hip_bf16.h>
#include <math.h>

typedef __attribute__((ext_vector_type(16))) __bf16 v16bf;
typedef __attribute__((ext_vector_type(8)))  float  v8f;

#define ETILE 128
#define NTHR  256

// Fast branchless tanh: prefer gfx1250 hardware v_tanh_f32 (TRANS op),
// else exp2/rcp-based approximation with clamp (no exec-mask branches).
static __device__ __forceinline__ float fast_tanh(float x) {
#if __has_builtin(__builtin_amdgcn_tanhf)
    return __builtin_amdgcn_tanhf(x);
#else
    // tanh(x) = 1 - 2/(exp(2x)+1); clamp so exp2 never overflows.
    float xc = fminf(fmaxf(x, -10.0f), 10.0f);
    float e  = __builtin_amdgcn_exp2f(xc * 2.8853900817779268f);  // 2*log2(e)
    float r  = __builtin_amdgcn_rcpf(e + 1.0f);
    return 1.0f - 2.0f * r;
#endif
}

// ---------------- small utility kernels ----------------

__global__ void k_fill(float* __restrict__ p, float v, int n) {
    int i = blockIdx.x * NTHR + threadIdx.x;
    if (i < n) p[i] = v;
}

__global__ void k_count(const int* __restrict__ J, float* __restrict__ deg, int E) {
    int e = blockIdx.x * NTHR + threadIdx.x;
    if (e < E) atomicAdd(&deg[J[e]], 1.0f);
}

__global__ void k_dinv(float* __restrict__ d, int n) {
    int i = blockIdx.x * NTHR + threadIdx.x;
    if (i < n) d[i] = rsqrtf(d[i]);           // deg >= 1 always (self loop)
}

__global__ void k_wcalc(const int* __restrict__ I, const int* __restrict__ J,
                        const float* __restrict__ dinv, float* __restrict__ W, int E) {
    int e = blockIdx.x * NTHR + threadIdx.x;
    if (e < E) W[e] = dinv[I[e]] * dinv[J[e]];
}

__global__ void k_convK(const float* __restrict__ K, __bf16* __restrict__ Kb, int n) {
    int i = blockIdx.x * NTHR + threadIdx.x;
    if (i < n) Kb[i] = (__bf16)K[i];
}

// x_new = 2*x_cur - x_old - H^2*dxn ; also re-zero dxn for next layer
__global__ void k_update(float4* __restrict__ xnw, const float4* __restrict__ xc,
                         const float4* __restrict__ xo, float4* __restrict__ dx, int n4) {
    int i = blockIdx.x * NTHR + threadIdx.x;
    if (i >= n4) return;
    const float h2 = 0.01f;   // H*H
    float4 c = xc[i], o = xo[i], d = dx[i], r;
    r.x = 2.f*c.x - o.x - h2*d.x;
    r.y = 2.f*c.y - o.y - h2*d.y;
    r.z = 2.f*c.z - o.z - h2*d.z;
    r.w = 2.f*c.w - o.w - h2*d.w;
    xnw[i] = r;
    float4 z = {0.f,0.f,0.f,0.f};
    dx[i] = z;
}

// ---------------- opening GEMM: x = relu(K1[64x128] @ xn[128xN]), node-major out
__global__ __launch_bounds__(NTHR) void k_open(const float* __restrict__ xn,
                                               const float* __restrict__ K1,
                                               float* __restrict__ x0,
                                               float* __restrict__ x1, int N) {
    __shared__ float K1s[64 * 128];
    for (int t = threadIdx.x; t < 64 * 128; t += NTHR) K1s[t] = K1[t];
    __syncthreads();
    int n = blockIdx.x * NTHR + threadIdx.x;
    if (n >= N) return;
    float acc[64];
#pragma unroll
    for (int o = 0; o < 64; ++o) acc[o] = 0.f;
    for (int c = 0; c < 128; ++c) {
        float v = xn[(size_t)c * N + n];
#pragma unroll
        for (int o = 0; o < 64; ++o) acc[o] = fmaf(K1s[o * 128 + c], v, acc[o]);
    }
#pragma unroll
    for (int o = 0; o < 64; ++o) {
        float r = acc[o] > 0.f ? acc[o] : 0.f;
        x0[(size_t)n * 64 + o] = r;
        x1[(size_t)n * 64 + o] = r;
    }
}

// ---------------- closing GEMM: out[n][o] = sum_c KNclose[o][c] * x[n][c]
__global__ __launch_bounds__(NTHR) void k_close(const float* __restrict__ x,
                                                const float* __restrict__ Kc,
                                                float* __restrict__ out, int N) {
    __shared__ float Ks[40 * 64];
    for (int t = threadIdx.x; t < 40 * 64; t += NTHR) Ks[t] = Kc[t];
    __syncthreads();
    int n = blockIdx.x * NTHR + threadIdx.x;
    if (n >= N) return;
    float xv[64];
    const float4* X4 = (const float4*)(x + (size_t)n * 64);
#pragma unroll
    for (int q = 0; q < 16; ++q) {
        float4 t = X4[q];
        xv[4*q] = t.x; xv[4*q+1] = t.y; xv[4*q+2] = t.z; xv[4*q+3] = t.w;
    }
    for (int o = 0; o < 40; ++o) {
        float a = 0.f;
#pragma unroll
        for (int c = 0; c < 64; ++c) a = fmaf(Ks[o * 64 + c], xv[c], a);
        out[(size_t)n * 40 + o] = a;
    }
}

// ---------------- main per-layer edge kernel (bf16 WMMA) ----------------
// x,dxn node-major [N][64].  Per block: 128 edges, 256 threads (8 waves).
// Wave w owns edge tile [w*16 .. w*16+15].
__global__ __launch_bounds__(NTHR) void k_edge(const float* __restrict__ x,
                                               const int* __restrict__ I,
                                               const int* __restrict__ J,
                                               const float* __restrict__ W,
                                               const __bf16* __restrict__ Kl,
                                               float* __restrict__ dxn, int E) {
    __shared__ __bf16 Ks[64 * 64];          // layer matrix (used by BOTH gemms)
    __shared__ __bf16 G_s[ETILE * 68];      // edge-major, pad stride 68
    __shared__ __bf16 T_s[ETILE * 68];
    __shared__ int    Ie[ETILE];
    __shared__ int    Je[ETILE];
    __shared__ float  Wv[ETILE];

    const int tid  = threadIdx.x;
    const int base = blockIdx.x * ETILE;

    for (int t = tid; t < 64 * 64; t += NTHR) Ks[t] = Kl[t];
    if (tid < ETILE) {
        int e = base + tid;
        if (e < E) { Ie[tid] = I[e]; Je[tid] = J[e]; Wv[tid] = W[e]; }
        else       { Ie[tid] = 0;    Je[tid] = 0;    Wv[tid] = 0.f; }
    }
    __syncthreads();

    // gather: 16 lane-groups of 16, each lane pulls a float4 (4 channels) per edge
    {
        const int g  = tid >> 4;
        const int lg = tid & 15;
        const float4* X4 = (const float4*)x;
        for (int i = 0; i < ETILE / 16; ++i) {
            int el = g * (ETILE / 16) + i;
            int ii = Ie[el], jj = Je[el];
            float w = Wv[el];
            float4 a = X4[(size_t)ii * 16 + lg];
            float4 b = X4[(size_t)jj * 16 + lg];
            int o4 = el * 68 + lg * 4;
            G_s[o4 + 0] = (__bf16)(w * (a.x - b.x));
            G_s[o4 + 1] = (__bf16)(w * (a.y - b.y));
            G_s[o4 + 2] = (__bf16)(w * (a.z - b.z));
            G_s[o4 + 3] = (__bf16)(w * (a.w - b.w));
        }
    }
    __syncthreads();

    const int lane = tid & 31;
    const int half = lane >> 4;
    const int ln16 = lane & 15;
    const int e0   = (tid >> 5) * 16;       // this wave's edge tile

    // A fragments of K (16x32 bf16 tiles); identical for gemm1 and gemm2
    v16bf A[4][2];
#pragma unroll
    for (int m = 0; m < 4; ++m) {
#pragma unroll
        for (int kc = 0; kc < 2; ++kc) {
            int r = m * 16 + ln16;
#pragma unroll
            for (int v = 0; v < 8; ++v) {
                int k0 = kc * 32 + half * 16 + 2 * v;
                A[m][kc][2 * v]     = Ks[r * 64 + k0];
                A[m][kc][2 * v + 1] = Ks[r * 64 + k0 + 1];
            }
        }
    }

    // GEMM1: T = tanh(K * G)
    {
        v16bf B[2];
#pragma unroll
        for (int kc = 0; kc < 2; ++kc)
#pragma unroll
            for (int v = 0; v < 8; ++v) {
                int k0 = kc * 32 + half * 16 + 2 * v;
                B[kc][2 * v]     = G_s[(e0 + ln16) * 68 + k0];
                B[kc][2 * v + 1] = G_s[(e0 + ln16) * 68 + k0 + 1];
            }
#pragma unroll
        for (int m = 0; m < 4; ++m) {
            v8f acc = {};
            acc = __builtin_amdgcn_wmma_f32_16x16x32_bf16(false, A[m][0], false, B[0],
                                                          (short)0, acc, false, false);
            acc = __builtin_amdgcn_wmma_f32_16x16x32_bf16(false, A[m][1], false, B[1],
                                                          (short)0, acc, false, false);
            int ocol  = e0 + ln16;
            int obase = m * 16 + half * 8;
#pragma unroll
            for (int r = 0; r < 8; ++r)
                T_s[ocol * 68 + obase + r] = (__bf16)fast_tanh(acc[r]);
        }
    }
    __syncthreads();

    // GEMM2: D = K * T ; scatter dW = W*D with f32 global atomics (L2-resident)
    {
        v16bf B[2];
#pragma unroll
        for (int kc = 0; kc < 2; ++kc)
#pragma unroll
            for (int v = 0; v < 8; ++v) {
                int k0 = kc * 32 + half * 16 + 2 * v;
                B[kc][2 * v]     = T_s[(e0 + ln16) * 68 + k0];
                B[kc][2 * v + 1] = T_s[(e0 + ln16) * 68 + k0 + 1];
            }
        int   el = e0 + ln16;
        int   ii = Ie[el], jj = Je[el];
        float w  = Wv[el];
#pragma unroll
        for (int m = 0; m < 4; ++m) {
            v8f acc = {};
            acc = __builtin_amdgcn_wmma_f32_16x16x32_bf16(false, A[m][0], false, B[0],
                                                          (short)0, acc, false, false);
            acc = __builtin_amdgcn_wmma_f32_16x16x32_bf16(false, A[m][1], false, B[1],
                                                          (short)0, acc, false, false);
            int obase = m * 16 + half * 8;
#pragma unroll
            for (int r = 0; r < 8; ++r) {
                float dw = w * acc[r];
                atomicAdd(&dxn[(size_t)ii * 64 + obase + r],  dw);
                atomicAdd(&dxn[(size_t)jj * 64 + obase + r], -dw);
            }
        }
    }
}

// ---------------- host orchestration ----------------
extern "C" void kernel_launch(void* const* d_in, const int* in_sizes, int n_in,
                              void* d_out, int out_size, void* d_ws, size_t ws_size,
                              hipStream_t stream) {
    const float* xn  = (const float*)d_in[0];
    const int*   I   = (const int*)  d_in[1];
    const int*   J   = (const int*)  d_in[2];
    const float* K1  = (const float*)d_in[4];
    const float* KN2 = (const float*)d_in[5];
    const float* KNc = (const float*)d_in[6];
    const int N = in_sizes[0] / 128;
    const int E = in_sizes[1];
    float* out = (float*)d_out;

    char* ws = (char*)d_ws;
    size_t off = 0;
    auto carve = [&](size_t b) -> char* {
        char* p = ws + off;
        off = (off + b + 255) & ~(size_t)255;
        return p;
    };
    float*  deg = (float*) carve((size_t)N * 4);            // becomes dinv in-place
    float*  W   = (float*) carve((size_t)E * 4);
    float*  xA  = (float*) carve((size_t)N * 64 * 4);
    float*  xB  = (float*) carve((size_t)N * 64 * 4);
    float*  xC  = (float*) carve((size_t)N * 64 * 4);
    float*  dxn = (float*) carve((size_t)N * 64 * 4);
    __bf16* Kb  = (__bf16*)carve((size_t)8 * 64 * 64 * 2);

    k_fill <<<(N + NTHR - 1) / NTHR, NTHR, 0, stream>>>(deg, 1.0f, N);  // self loops
    k_count<<<(E + NTHR - 1) / NTHR, NTHR, 0, stream>>>(J, deg, E);
    k_dinv <<<(N + NTHR - 1) / NTHR, NTHR, 0, stream>>>(deg, N);
    k_wcalc<<<(E + NTHR - 1) / NTHR, NTHR, 0, stream>>>(I, J, deg, W, E);
    k_convK<<<(8 * 64 * 64 + NTHR - 1) / NTHR, NTHR, 0, stream>>>(KN2, Kb, 8 * 64 * 64);
    k_open <<<(N + NTHR - 1) / NTHR, NTHR, 0, stream>>>(xn, K1, xA, xB, N);
    k_fill <<<(N * 64 + NTHR - 1) / NTHR, NTHR, 0, stream>>>(dxn, 0.0f, N * 64);

    float* xc = xA; float* xo = xB; float* xnw = xC;
    const int n4 = N * 16;
    for (int l = 0; l < 8; ++l) {
        k_edge  <<<(E + ETILE - 1) / ETILE, NTHR, 0, stream>>>(xc, I, J, W,
                                                               Kb + (size_t)l * 4096, dxn, E);
        k_update<<<(n4 + NTHR - 1) / NTHR, NTHR, 0, stream>>>((float4*)xnw, (const float4*)xc,
                                                              (const float4*)xo, (float4*)dxn, n4);
        float* t = xo; xo = xc; xc = xnw; xnw = t;
    }
    k_close<<<(N + NTHR - 1) / NTHR, NTHR, 0, stream>>>(xc, KNc, out, N);
}
